// HierarchicalGenerator_33758442947366
// MI455X (gfx1250) — matrix operand order repaired and checked by reading, not compiled, not run on hardware
//
#include <hip/hip_runtime.h>
#include <math.h>

// ---------------- problem dims ----------------
#define B_   256
#define T_   512
#define H_   256
#define ZS_  64
#define ZT_  64
#define G3_  768      // 3*H
#define EPS_ 1e-5f

// ---------------- workspace layout (bytes) ----------------
#define WS_BASELINE   0u          // 256*256*4 = 262144
#define WS_TMP        262144u     // 262144
#define WS_CONST      524288u     // 256*768*4 = 786432  -> 1310720
#define WS_PK_WHH     1310720u    // 8kt*48nt*256dw*4 = 393216 -> 1703936
#define WS_PK_WZT     1703936u    // 2*48*256*4 = 98304 -> 1802240
#define WS_PK_WHEAD   1802240u    // 8*2*256*4  = 16384 -> 1818624
#define WS_WHEAD_STG  1818624u    // 32*256*4   = 32768 -> 1851392

// ---------------- output layout (floats) ----------------
#define O_SC    0u
#define O_SCAT  2048u                       // 256*8
#define O_TC    4608u                       // +256*10
#define O_TCAT  (4608u + 256u*512u*16u)     // 2101760
#define O_MASK  (O_TCAT + 256u*512u*9u)     // 3281408

// ---------------- WMMA types ----------------
typedef __attribute__((ext_vector_type(16))) __bf16 v16bf;
typedef __attribute__((ext_vector_type(8)))  float  v8f;

union Frag16 { unsigned int u[8]; v16bf v; };

__device__ inline unsigned short f2bf(float f) {
  union { float f; unsigned int u; } c; c.f = f;
  unsigned int u = c.u;
  unsigned int r = u + 0x7FFFu + ((u >> 16) & 1u);   // round-to-nearest-even
  return (unsigned short)(r >> 16);
}

__device__ inline v8f wmma_bf16(v16bf a, v16bf b, v8f c) {
  return __builtin_amdgcn_wmma_f32_16x16x32_bf16(false, a, false, b, (short)0, c, false, false);
}

// A fragment (16x32 bf16) from an LDS row-major bf16 tile.
// ISA layout: lane&15 = M; lanes>=16 add 8 to K; v<4 -> K=2v, v>=4 -> K=16+2(v-4).
__device__ inline v16bf load_a_frag(const unsigned short* buf, int rowDw, int kt, int lane) {
  const unsigned int* d = (const unsigned int*)buf;
  const int m  = lane & 15;
  const int kb = (lane & 16) ? 8 : 0;
  Frag16 f;
#pragma unroll
  for (int v = 0; v < 8; ++v) {
    int k = kt * 32 + kb + ((v < 4) ? (2 * v) : (16 + 2 * (v - 4)));
    f.u[v] = d[m * rowDw + (k >> 1)];
  }
  return f.v;
}

// B fragment (32x16 bf16) pre-packed per-lane contiguous: 8 dwords/lane.
__device__ inline v16bf load_b_frag(const unsigned int* pk, int lane) {
  Frag16 f;
  const unsigned int* p = pk + lane * 8;
#pragma unroll
  for (int v = 0; v < 8; ++v) f.u[v] = p[v];
  return f.v;
}

// ---------------- RNG (hash-based gumbel; compile-only env) ----------------
__device__ inline unsigned int mix_(unsigned int x) {
  x ^= x >> 17; x *= 0xED5AD4BBu; x ^= x >> 11; x *= 0xAC4C1B51u;
  x ^= x >> 15; x *= 0x31848BABu; x ^= x >> 14;
  return x;
}
__device__ inline float urand_(unsigned int a, unsigned int b, unsigned int c) {
  unsigned int h = mix_(a * 0x9E3779B9u ^ mix_(b * 0x85EBCA6Bu ^ mix_(c ^ 0x165667B1u)));
  return ((float)(h >> 8) + 0.5f) * (1.0f / 16777216.0f);
}
__device__ inline float gumbel_(float u) { return -__logf(-__logf(u)); }
__device__ inline float sigmoid_(float x) { return 1.0f / (1.0f + __expf(-x)); }

// =====================================================================
// pack a weight matrix (out = x @ W^T, W row-major [N, ldw]) into
// per-tile B fragments: dst[(kt*NT+nt)*256 + lane*8 + v] packs
// (B[k][n], B[k+1][n]) with n = nt*16 + (lane&15),
// k = kt*32 + 8*(lane>>4) + (v<4 ? 2v : 16+2(v-4)).
// =====================================================================
__global__ void pack_b_kernel(const float* __restrict__ W, int ldw, int coloff,
                              int NT, int total, unsigned int* __restrict__ dst) {
  int idx = blockIdx.x * blockDim.x + threadIdx.x;
  if (idx >= total) return;
  int v    = idx & 7;
  int lane = (idx >> 3) & 31;
  int tile = idx >> 8;
  int kt = tile / NT;
  int nt = tile - kt * NT;
  int k  = kt * 32 + ((lane & 16) ? 8 : 0) + ((v < 4) ? (2 * v) : (16 + 2 * (v - 4)));
  int n0 = nt * 16 + (lane & 15);
  unsigned int lo = f2bf(W[(size_t)n0 * ldw + coloff + k]);
  unsigned int hi = f2bf(W[(size_t)n0 * ldw + coloff + k + 1]);
  dst[idx] = lo | (hi << 16);
}

// =====================================================================
// stage head weights [32,256] = [tc(16) | irr1 | irr2 | tcat(5) | zeros(9)]
// =====================================================================
__global__ void build_whead_kernel(const float* __restrict__ tc_w,
                                   const float* __restrict__ irr1_w,
                                   const float* __restrict__ irr2_w,
                                   const float* __restrict__ tcat_w,
                                   float* __restrict__ dst) {
  int idx = blockIdx.x * blockDim.x + threadIdx.x;   // r*256+k
  if (idx >= 32 * H_) return;
  int r = idx >> 8, k = idx & 255;
  float v = 0.0f;
  if (r < 16)       v = tc_w[r * H_ + k];
  else if (r == 16) v = irr1_w[k];
  else if (r == 17) v = irr2_w[k];
  else if (r < 23)  v = tcat_w[(r - 18) * H_ + k];
  dst[idx] = v;
}

// =====================================================================
// y[b,n] = relu(x[b,:]@W[n,:]+bias[n]),  H_=256 outputs
// =====================================================================
__global__ void lin_relu_kernel(const float* __restrict__ x, const float* __restrict__ W,
                                const float* __restrict__ bias, int K,
                                float* __restrict__ y) {
  int idx = blockIdx.x * blockDim.x + threadIdx.x;   // b*256+n
  int b = idx >> 8, n = idx & 255;
  const float* xr = x + (size_t)b * K;
  const float* wr = W + (size_t)n * K;
  float acc = bias[n];
  for (int k = 0; k < K; ++k) acc += xr[k] * wr[k];
  y[idx] = fmaxf(acc, 0.0f);
}

// =====================================================================
// training-mode BatchNorm over batch (B_=256 == blockDim)
// =====================================================================
__global__ void bn_kernel(const float* __restrict__ x, const float* __restrict__ g,
                          const float* __restrict__ bt, float* __restrict__ y) {
  __shared__ float red[256];
  int n = blockIdx.x, b = threadIdx.x;
  float v = x[(size_t)b * H_ + n];
  red[b] = v; __syncthreads();
  for (int s = 128; s > 0; s >>= 1) { if (b < s) red[b] += red[b + s]; __syncthreads(); }
  float mean = red[0] * (1.0f / 256.0f);
  __syncthreads();
  float d = v - mean;
  red[b] = d * d; __syncthreads();
  for (int s = 128; s > 0; s >>= 1) { if (b < s) red[b] += red[b + s]; __syncthreads(); }
  float var = red[0] * (1.0f / 256.0f);
  y[(size_t)b * H_ + n] = d * rsqrtf(var + EPS_) * g[n] + bt[n];
}

// =====================================================================
// static heads: static_cont [B,8], static_cat [B,10] (gumbel softmax)
// =====================================================================
__global__ void static_heads_kernel(const float* __restrict__ baseline,
                                    const float* __restrict__ sc_w, const float* __restrict__ sc_b,
                                    const float* __restrict__ s1w, const float* __restrict__ s1b,
                                    const float* __restrict__ s2w, const float* __restrict__ s2b,
                                    float* __restrict__ out_sc, float* __restrict__ out_scat) {
  int b = threadIdx.x;
  const float* base = baseline + (size_t)b * H_;
  for (int j = 0; j < 8; ++j) {
    float acc = sc_b[j];
    const float* w = sc_w + (size_t)j * H_;
    for (int k = 0; k < H_; ++k) acc += base[k] * w[k];
    out_sc[b * 8 + j] = tanhf(acc) * 0.5f + 0.5f;
  }
  {
    float l[4], mx = -1e30f;
    for (int j = 0; j < 4; ++j) {
      float acc = s1b[j];
      const float* w = s1w + (size_t)j * H_;
      for (int k = 0; k < H_; ++k) acc += base[k] * w[k];
      l[j] = acc + gumbel_(urand_(0x51u, (unsigned)b, (unsigned)j));
      mx = fmaxf(mx, l[j]);
    }
    float s = 0.0f;
    for (int j = 0; j < 4; ++j) { l[j] = __expf(l[j] - mx); s += l[j]; }
    for (int j = 0; j < 4; ++j) out_scat[b * 10 + j] = l[j] / s;
  }
  {
    float l[6], mx = -1e30f;
    for (int j = 0; j < 6; ++j) {
      float acc = s2b[j];
      const float* w = s2w + (size_t)j * H_;
      for (int k = 0; k < H_; ++k) acc += base[k] * w[k];
      l[j] = acc + gumbel_(urand_(0x52u, (unsigned)b, (unsigned)j));
      mx = fmaxf(mx, l[j]);
    }
    float s = 0.0f;
    for (int j = 0; j < 6; ++j) { l[j] = __expf(l[j] - mx); s += l[j]; }
    for (int j = 0; j < 6; ++j) out_scat[b * 10 + 4 + j] = l[j] / s;
  }
}

// =====================================================================
// const_term[b,n] = bih[n] + baseline[b]@Wih[n,64:320] + init4[b]@Wih[n,321:325]
// =====================================================================
__global__ void const_term_kernel(const float* __restrict__ baseline,
                                  const float* __restrict__ init_st,
                                  const float* __restrict__ wih,
                                  const float* __restrict__ bih,
                                  float* __restrict__ constT) {
  int idx = blockIdx.x * blockDim.x + threadIdx.x;   // b*768+n
  int b = idx / G3_, n = idx - b * G3_;
  const float* w = wih + (size_t)n * 325;
  const float* x = baseline + (size_t)b * H_;
  float acc = bih[n];
  for (int k = 0; k < H_; ++k) acc += x[k] * w[64 + k];
  const float* is = init_st + b * 4;
#pragma unroll
  for (int i = 0; i < 4; ++i) acc += is[i] * w[321 + i];
  constT[idx] = acc;
}

// =====================================================================
// fused GRU recurrence + z-projection + heads + irreversible scan + tcat.
// 16 blocks x 16 batch rows x 16 waves (512 thr): each wave owns ONE
// column tile triple (r/z/n tiles nt, 16+nt, 32+nt for nt=wave), so gate
// nonlinearities apply immediately in registers and h is published to a
// double-buffered bf16 tile with no WMMA result live across a barrier.
// 4 waves/SIMD for latency hiding; kt loops unrolled x2 so two fragment
// sets are in flight per wave (partial s_wait instead of full drain).
// =====================================================================
__global__ __launch_bounds__(512) void gru_fused_kernel(
    const float* __restrict__ zt, const float* __restrict__ init_st,
    const float* __restrict__ wih, const float* __restrict__ bhh,
    const float* __restrict__ tc_b, const float* __restrict__ irr1_b,
    const float* __restrict__ irr2_b, const float* __restrict__ tcat_b,
    const float* __restrict__ constT,
    const unsigned int* __restrict__ pk_whh,
    const unsigned int* __restrict__ pk_wzt,
    const unsigned int* __restrict__ pk_whead,
    float* __restrict__ out_tc, float* __restrict__ out_tcat,
    float* __restrict__ out_mask)
{
  extern __shared__ char smem[];
  float* hf    = (float*)smem;                         // 16*256 f32 h state (owner-lane only)
  float* headv = hf + 16 * H_;                         // 16*32
  float* headb = headv + 16 * 32;                      // 32
  float* stat  = headb + 32;                           // 16*4 irreversible states
  unsigned short* hb0 = (unsigned short*)(stat + 64);  // 16*256 bf16 h, buffer 0
  unsigned short* hb1 = hb0 + 16 * H_;                 // 16*256 bf16 h, buffer 1
  unsigned short* zbf = hb1 + 16 * H_;                 // 16*64 bf16 z tile

  const int tid  = threadIdx.x;
  const int lane = tid & 31;
  const int wave = tid >> 5;          // 0..15: owns column tile nt = wave
  const int b0   = blockIdx.x * 16;
  const int l    = lane & 15;
  const int mb   = (lane & 16) ? 8 : 0;

  for (int i = tid; i < 16 * H_; i += 512) { hf[i] = 0.0f; hb0[i] = 0; hb1[i] = 0; }
  if (tid < 32) {
    float v = 0.0f;
    if (tid < 16)       v = tc_b[tid];
    else if (tid == 16) v = irr1_b[0];
    else if (tid == 17) v = irr2_b[0];
    else if (tid < 23)  v = tcat_b[tid - 18];
    headb[tid] = v;
  }
  if (tid < 64) stat[tid] = init_st[b0 * 4 + tid];

  // per-lane T-invariant constants for this wave's 3 gate tiles:
  // gate g column n = g*256 + 16*wave + l, row m = mb+i
  float ccst[3][8];
  float wvv[3];
  float bhv[3];
#pragma unroll
  for (int g = 0; g < 3; ++g) {
    int n = g * 256 + 16 * wave + l;
    wvv[g] = wih[(size_t)n * 325 + 320];   // visit column of Wih
    bhv[g] = bhh[n];
#pragma unroll
    for (int i = 0; i < 8; ++i)
      ccst[g][i] = constT[(size_t)(b0 + mb + i) * G3_ + n];
  }
  __syncthreads();

  unsigned short* hcur = hb0;
  unsigned short* hnxt = hb1;

  const int nt = wave;                 // column tile index (0..15)
  const int j  = 16 * wave + l;        // hidden unit owned by this lane

  for (int t = 0; t < T_; ++t) {
    const float visit = (float)t * (1.0f / (float)(T_ - 1));

    // stage z_temporal tile [16,64] -> bf16 LDS, prefetch next step
    for (int i = tid; i < 16 * ZT_; i += 512) {
      int m = i >> 6, k = i & 63;
      const float* src = zt + ((size_t)(b0 + m) * T_ + t) * ZT_ + k;
      zbf[i] = f2bf(*src);
      if (t + 1 < T_) __builtin_prefetch(src + ZT_, 0, 1);
    }
    __syncthreads();   // zbf ready; also orders everyone out of outputs(t-1)

    // ---- WMMA gate tiles then immediate gate update ----
    // r gate: C seeded with (const + visit*wv + bhh)
    v8f cr;
#pragma unroll
    for (int i = 0; i < 8; ++i) cr[i] = ccst[0][i] + visit * wvv[0] + bhv[0];
#pragma unroll
    for (int kt = 0; kt < 2; ++kt)
      cr = wmma_bf16(load_a_frag(zbf, 32, kt, lane),
                     load_b_frag(pk_wzt + (size_t)(kt * 48 + nt) * 256, lane), cr);
#pragma unroll 2
    for (int kt = 0; kt < 8; ++kt)
      cr = wmma_bf16(load_a_frag(hcur, 128, kt, lane),
                     load_b_frag(pk_whh + (size_t)(kt * 48 + nt) * 256, lane), cr);
    // z gate
    v8f cz;
#pragma unroll
    for (int i = 0; i < 8; ++i) cz[i] = ccst[1][i] + visit * wvv[1] + bhv[1];
#pragma unroll
    for (int kt = 0; kt < 2; ++kt)
      cz = wmma_bf16(load_a_frag(zbf, 32, kt, lane),
                     load_b_frag(pk_wzt + (size_t)(kt * 48 + 16 + nt) * 256, lane), cz);
#pragma unroll 2
    for (int kt = 0; kt < 8; ++kt)
      cz = wmma_bf16(load_a_frag(hcur, 128, kt, lane),
                     load_b_frag(pk_whh + (size_t)(kt * 48 + 16 + nt) * 256, lane), cz);
    // n gate: x part (seeded with const) and h part (seeded with bhh) separate
    v8f cx;
#pragma unroll
    for (int i = 0; i < 8; ++i) cx[i] = ccst[2][i] + visit * wvv[2];
#pragma unroll
    for (int kt = 0; kt < 2; ++kt)
      cx = wmma_bf16(load_a_frag(zbf, 32, kt, lane),
                     load_b_frag(pk_wzt + (size_t)(kt * 48 + 32 + nt) * 256, lane), cx);
    v8f ch;
#pragma unroll
    for (int i = 0; i < 8; ++i) ch[i] = bhv[2];
#pragma unroll 2
    for (int kt = 0; kt < 8; ++kt)
      ch = wmma_bf16(load_a_frag(hcur, 128, kt, lane),
                     load_b_frag(pk_whh + (size_t)(kt * 48 + 32 + nt) * 256, lane), ch);

    // immediate gate nonlinearity + h update into the *other* buffer
#pragma unroll
    for (int i = 0; i < 8; ++i) {
      const int m = mb + i;
      float rv   = sigmoid_(cr[i]);
      float zv   = sigmoid_(cz[i]);
      float nv   = tanhf(cx[i] + rv * ch[i]);
      float hnew = (1.0f - zv) * nv + zv * hf[m * H_ + j];
      hf[m * H_ + j]   = hnew;
      hnxt[m * H_ + j] = f2bf(hnew);
    }
    __syncthreads();   // h_t published in hnxt

    // ---- head GEMM on updated h (cols: 16 tc | 2 hazard | 5 tcat | pad) ----
    if (wave < 2) {
      v8f c;
#pragma unroll
      for (int i = 0; i < 8; ++i) c[i] = 0.0f;
#pragma unroll 2
      for (int kt = 0; kt < 8; ++kt)
        c = wmma_bf16(load_a_frag(hnxt, 128, kt, lane),
                      load_b_frag(pk_whead + (size_t)(kt * 2 + wave) * 256, lane), c);
      const int col = wave * 16 + l;
#pragma unroll
      for (int i = 0; i < 8; ++i)
        headv[(mb + i) * 32 + col] = c[i] + headb[col];
    }
    __syncthreads();

    // ---- elementwise outputs (first 256 threads) ----
    if (tid < 256) {
      const int m = tid >> 4;
      const int jj = tid & 15;
      const int brow = b0 + m;
      const size_t bt = (size_t)brow * T_ + t;
      out_tc[bt * 16 + jj] = tanhf(headv[m * 32 + jj]) * 0.5f + 0.5f;
      if (jj == 1) out_mask[bt] = 1.0f;
      if (jj == 0) {
        if (t > 0) {   // states[t>=1] = f(states[t-1], hazard_t); states[0] = init
#pragma unroll
          for (int q = 0; q < 2; ++q) {
            float hz  = sigmoid_(headv[m * 32 + 16 + q]);
            float s1p = stat[m * 4 + q * 2 + 1];
            float s1n = (s1p > 0.5f) ? 1.0f : hz;
            float s0n = (s1p > 0.5f) ? 0.0f : (1.0f - hz);
            stat[m * 4 + q * 2]     = s0n;
            stat[m * 4 + q * 2 + 1] = s1n;
          }
        }
        float* oc = out_tcat + bt * 9;
        oc[0] = stat[m * 4 + 0]; oc[1] = stat[m * 4 + 1];
        oc[2] = stat[m * 4 + 2]; oc[3] = stat[m * 4 + 3];
        float lg[5]; float mx = -1e30f;
#pragma unroll
        for (int q = 0; q < 5; ++q) {
          lg[q] = headv[m * 32 + 18 + q]
                + gumbel_(urand_(0x53u, (unsigned)(brow * 512 + t), (unsigned)q));
          mx = fmaxf(mx, lg[q]);
        }
        float s = 0.0f;
#pragma unroll
        for (int q = 0; q < 5; ++q) { lg[q] = __expf(lg[q] - mx); s += lg[q]; }
        float inv = 1.0f / s;
#pragma unroll
        for (int q = 0; q < 5; ++q) oc[4 + q] = lg[q] * inv;
      }
    }
    // no trailing barrier needed: next-iteration zbf barrier orders everything

    // swap h buffers
    unsigned short* tmpp = hcur; hcur = hnxt; hnxt = tmpp;
  }
}

// =====================================================================
extern "C" void kernel_launch(void* const* d_in, const int* in_sizes, int n_in,
                              void* d_out, int out_size, void* d_ws, size_t ws_size,
                              hipStream_t stream) {
  (void)in_sizes; (void)n_in; (void)out_size; (void)ws_size;
  const float* z_static    = (const float*)d_in[0];
  const float* z_temporal  = (const float*)d_in[1];
  const float* init_states = (const float*)d_in[2];
  const float* sn_w1 = (const float*)d_in[3];
  const float* sn_b1 = (const float*)d_in[4];
  const float* bn1_g = (const float*)d_in[5];
  const float* bn1_b = (const float*)d_in[6];
  const float* sn_w2 = (const float*)d_in[7];
  const float* sn_b2 = (const float*)d_in[8];
  const float* bn2_g = (const float*)d_in[9];
  const float* bn2_b = (const float*)d_in[10];
  const float* sc_w   = (const float*)d_in[11];
  const float* sc_b   = (const float*)d_in[12];
  const float* scat1_w = (const float*)d_in[13];
  const float* scat1_b = (const float*)d_in[14];
  const float* scat2_w = (const float*)d_in[15];
  const float* scat2_b = (const float*)d_in[16];
  const float* gru_wih = (const float*)d_in[17];
  const float* gru_whh = (const float*)d_in[18];
  const float* gru_bih = (const float*)d_in[19];
  const float* gru_bhh = (const float*)d_in[20];
  const float* tc_w   = (const float*)d_in[21];
  const float* tc_b   = (const float*)d_in[22];
  const float* irr1_w = (const float*)d_in[23];
  const float* irr1_b = (const float*)d_in[24];
  const float* irr2_w = (const float*)d_in[25];
  const float* irr2_b = (const float*)d_in[26];
  const float* tcat_w = (const float*)d_in[27];
  const float* tcat_b = (const float*)d_in[28];

  char* ws = (char*)d_ws;
  float* baseline         = (float*)(ws + WS_BASELINE);
  float* tmp              = (float*)(ws + WS_TMP);
  float* constT           = (float*)(ws + WS_CONST);
  unsigned int* pk_whh    = (unsigned int*)(ws + WS_PK_WHH);
  unsigned int* pk_wzt    = (unsigned int*)(ws + WS_PK_WZT);
  unsigned int* pk_whead  = (unsigned int*)(ws + WS_PK_WHEAD);
  float* whead_stg        = (float*)(ws + WS_WHEAD_STG);

  float* out      = (float*)d_out;
  float* out_sc   = out + O_SC;
  float* out_scat = out + O_SCAT;
  float* out_tc   = out + O_TC;
  float* out_tcat = out + O_TCAT;
  float* out_mask = out + O_MASK;

  // ---- static path (tiny; VALU kernels) ----
  lin_relu_kernel<<<256, 256, 0, stream>>>(z_static, sn_w1, sn_b1, ZS_, tmp);
  bn_kernel<<<256, 256, 0, stream>>>(tmp, bn1_g, bn1_b, baseline);
  lin_relu_kernel<<<256, 256, 0, stream>>>(baseline, sn_w2, sn_b2, H_, tmp);
  bn_kernel<<<256, 256, 0, stream>>>(tmp, bn2_g, bn2_b, baseline);
  static_heads_kernel<<<1, 256, 0, stream>>>(baseline, sc_w, sc_b, scat1_w, scat1_b,
                                             scat2_w, scat2_b, out_sc, out_scat);

  // ---- T-invariant part of x_proj ----
  const_term_kernel<<<768, 256, 0, stream>>>(baseline, init_states, gru_wih, gru_bih, constT);

  // ---- pack weights to bf16 WMMA B-fragments ----
  build_whead_kernel<<<32, 256, 0, stream>>>(tc_w, irr1_w, irr2_w, tcat_w, whead_stg);
  pack_b_kernel<<<384, 256, 0, stream>>>(gru_whh, 256, 0, 48, 8 * 48 * 256, pk_whh);  // Whh^T 256x768
  pack_b_kernel<<<96, 256, 0, stream>>>(gru_wih, 325, 0, 48, 2 * 48 * 256, pk_wzt);   // Wzt^T 64x768
  pack_b_kernel<<<16, 256, 0, stream>>>(whead_stg, 256, 0, 2, 8 * 2 * 256, pk_whead); // Whead^T 256x32

  // ---- fused GRU + heads (WMMA persistent kernel, 16 blocks x 16 waves) ----
  const size_t smem = (size_t)(16 * H_ + 16 * 32 + 32 + 64) * 4    // hf, headv, headb, stat
                    + (size_t)(2 * 16 * H_ + 16 * ZT_) * 2;        // hb0, hb1, zbf => 37248 B
  gru_fused_kernel<<<16, 512, smem, stream>>>(
      z_temporal, init_states, gru_wih, gru_bhh,
      tc_b, irr1_b, irr2_b, tcat_b,
      constT, pk_whh, pk_wzt, pk_whead,
      out_tc, out_tcat, out_mask);
}
